// GroupedQueryAttention_63926293234009
// MI455X (gfx1250) — matrix operand order repaired and checked by reading, not compiled, not run on hardware
//
#include <hip/hip_runtime.h>

// ---------------- problem constants ----------------
constexpr int B_   = 2;
constexpr int S_   = 2048;
constexpr int D_   = 4096;
constexpr int NQ_  = 32;
constexpr int NKV_ = 4;
constexpr int DK_  = 128;
constexpr int GROUPS_ = NQ_ / NKV_;   // 8
constexpr int ROWS_ = B_ * S_;        // 4096

typedef __bf16 bf16;
typedef __attribute__((ext_vector_type(16))) __bf16 v16bf;
typedef __attribute__((ext_vector_type(8)))  float  v8f;
typedef __attribute__((ext_vector_type(2)))  __bf16 bf16x2;
typedef __attribute__((ext_vector_type(4)))  unsigned int u32x4;
typedef __attribute__((ext_vector_type(8)))  unsigned int u32x8;

union AFrag { v16bf v; bf16x2 p[8]; };
union CFrag { v8f   v; float  e[8]; };

__device__ __forceinline__ bf16 to_bf16(float x) { return (bf16)x; }
__device__ __forceinline__ bf16 to_bf16(bf16  x) { return x; }
__device__ __forceinline__ void store_c(float* p, float v) { *p = v; }
__device__ __forceinline__ void store_c(bf16*  p, float v) { *p = (bf16)v; }

// LDS byte offset of a __shared__ object: flat-LDS aperture keeps the
// group-segment offset in addr[31:0].
__device__ __forceinline__ unsigned lds_off(const void* p) {
  return (unsigned)(size_t)p;
}

// Issue one TDM K-tile + one TDM V-tile load (2D 128x64 bf16 tiles).
// Group-1 descriptor (geometry) is shared; group-0 carries addresses.
__device__ __forceinline__ void tdm_issue_pair(const bf16* kp, const bf16* vp,
                                               unsigned ldsK, unsigned ldsV,
                                               u32x8 g1) {
  unsigned long long kga = (unsigned long long)(size_t)kp;
  unsigned long long vga = (unsigned long long)(size_t)vp;
  u32x4 g0k, g0v;
  g0k[0] = 1u;                                     // count=1 (valid user D#)
  g0k[1] = ldsK;                                   // lds_addr
  g0k[2] = (unsigned)(kga & 0xffffffffu);          // global_addr[31:0]
  g0k[3] = (unsigned)((kga >> 32) & 0x01ffffffu)   // global_addr[56:32]
           | 0x80000000u;                          // type=2 ("image")
  g0v[0] = 1u;
  g0v[1] = ldsV;
  g0v[2] = (unsigned)(vga & 0xffffffffu);
  g0v[3] = (unsigned)((vga >> 32) & 0x01ffffffu) | 0x80000000u;
  asm volatile("tensor_load_to_lds %0, %1" :: "s"(g0k), "s"(g1) : "memory");
  asm volatile("tensor_load_to_lds %0, %1" :: "s"(g0v), "s"(g1) : "memory");
}

// =====================================================================
// Tiled WMMA GEMM: C[M,N] = A[M,K] * B[K,N], row-major.
// Block tile 128x128, Kc=32, 8 waves (2x4), each wave -> 64x32 of C.
// A/B converted to bf16 while staging into LDS; WMMA bf16 -> f32.
// =====================================================================
template <typename TA, typename TB, typename TC>
__global__ __launch_bounds__(256)
void gemm_wmma_bf16(const TA* __restrict__ A, const TB* __restrict__ Bm,
                    TC* __restrict__ C, int M, int N, int K) {
  __shared__ bf16 As[128][34];   // [m][k], padded
  __shared__ bf16 Bs[128][34];   // [n][k] (transposed), padded

  const int t    = threadIdx.x;
  const int w    = t >> 5;
  const int lane = t & 31;
  const int lo   = lane & 15;
  const int hi   = lane >> 4;

  const int bm = blockIdx.y * 128;
  const int bn = blockIdx.x * 128;
  const int wm = (w >> 2) * 64;   // wave row offset in tile
  const int wn = (w & 3) * 32;    // wave col offset in tile

  CFrag acc[4][2] = {};

  for (int k0 = 0; k0 < K; k0 += 32) {
    __syncthreads();
    // stage A tile 128x32 (fp32 or bf16 -> bf16)
#pragma unroll
    for (int i = 0; i < 16; ++i) {
      int li = i * 256 + t;
      int r = li >> 5, c = li & 31;
      As[r][c] = to_bf16(A[(size_t)(bm + r) * K + (k0 + c)]);
    }
    // stage B tile 32x128 transposed -> Bs[n][k]
#pragma unroll
    for (int i = 0; i < 16; ++i) {
      int li = i * 256 + t;
      int r = li >> 7, c = li & 127;
      Bs[c][r] = to_bf16(Bm[(size_t)(k0 + r) * N + (bn + c)]);
    }
    if (k0 + 32 < K) {
      __builtin_prefetch(&A[(size_t)(bm + (t >> 1)) * K + (k0 + 32) + (t & 1) * 16], 0, 1);
      __builtin_prefetch(&Bm[(size_t)(k0 + 32 + (t >> 3)) * N + bn + ((t & 7) << 4)], 0, 1);
    }
    __syncthreads();

    // build fragments (same contiguous k order for A and B: k = 16*hi + s)
    AFrag afr[4], bfr[2];
#pragma unroll
    for (int tm = 0; tm < 4; ++tm)
#pragma unroll
      for (int v = 0; v < 8; ++v) {
        int kk = hi * 16 + 2 * v;                              // A: 16x32 layout
        afr[tm].p[v] = *(const bf16x2*)&As[wm + tm * 16 + lo][kk];
      }
#pragma unroll
    for (int tn = 0; tn < 2; ++tn)
#pragma unroll
      for (int v = 0; v < 8; ++v) {
        int kk = hi * 16 + 2 * v;                              // B: 32x16 layout
        bfr[tn].p[v] = *(const bf16x2*)&Bs[wn + tn * 16 + lo][kk];
      }
#pragma unroll
    for (int tm = 0; tm < 4; ++tm)
#pragma unroll
      for (int tn = 0; tn < 2; ++tn)
        acc[tm][tn].v = __builtin_amdgcn_wmma_f32_16x16x32_bf16(
            false, afr[tm].v, false, bfr[tn].v, (short)0, acc[tm][tn].v, false, false);
  }

  // write back: C tile layout -> m = r + 8*hi, n = lo
#pragma unroll
  for (int tm = 0; tm < 4; ++tm)
#pragma unroll
    for (int tn = 0; tn < 2; ++tn)
#pragma unroll
      for (int r = 0; r < 8; ++r) {
        int row = bm + wm + tm * 16 + hi * 8 + r;
        int col = bn + wn + tn * 16 + lo;
        store_c(&C[(size_t)row * N + col], acc[tm][tn].e[r]);
      }
}

// =====================================================================
// Causal GQA flash attention over bf16 Q/K/V (fp32 accumulation).
// grid: x = S/64 query blocks, y = B*NQ. block = 128 threads (4 waves),
// each wave owns 16 query rows, full DK=128 accumulator (8 C tiles).
// Q tile staged via async global->LDS copies (ASYNCcnt path).
// K/V tiles double-buffered through the Tensor Data Mover: the next
// tile's DMA runs concurrently with WMMA on the current tile
// (s_wait_tensorcnt <= 2 drains only the older K/V pair).
// =====================================================================
__global__ __launch_bounds__(128)
void gqa_flash_attn(const bf16* __restrict__ Q, const bf16* __restrict__ Kg,
                    const bf16* __restrict__ Vg, bf16* __restrict__ O) {
  __shared__ bf16 Qs[64][128];        // [row][dk]
  __shared__ bf16 Ks[2][64][128];     // [buf][key][dk]   (TDM destination)
  __shared__ bf16 Vs[2][64][128];     // [buf][key][dk]   (TDM destination)
  __shared__ bf16 VT[128][72];        // [dk][key], padded (PV B-fragments)
  __shared__ bf16 Ps[4][16][64];      // per-wave P staging [row][key]

  const int t    = threadIdx.x;
  const int w    = t >> 5;
  const int lane = t & 31;
  const int lo   = lane & 15;
  const int hi   = lane >> 4;

  const int qb  = blockIdx.x;          // 64-row query block
  const int bh  = blockIdx.y;
  const int bb  = bh / NQ_;
  const int h   = bh % NQ_;
  const int kvh = h / GROUPS_;

  const float scale = 0.08838834764831845f;  // 1/sqrt(128)

  // TDM descriptor group 1 (identical geometry for K and V tensors):
  // data_size=2B, tensor_dim0=512, tensor_dim1=4096, tile 128x64, stride0=512.
  u32x8 g1;
  g1[0] = 0x00010000u;          // data_size=1 (2 bytes), no multicast/pad/iterate
  g1[1] = (512u << 16);         // tensor_dim0[15:0]=512 in bits[63:48]
  g1[2] = (4096u << 16);        // tensor_dim0 hi=0, tensor_dim1[15:0]=4096
  g1[3] = (128u << 16);         // tensor_dim1 hi=0, tile_dim0=128
  g1[4] = 64u;                  // tile_dim1=64, tile_dim2=0
  g1[5] = 512u;                 // tensor_dim0_stride lo = 512
  g1[6] = 0u;                   // stride hi / tensor_dim1_stride lo
  g1[7] = 0u;

  const size_t kvstride = (size_t)(NKV_ * DK_);
  const bf16* kbase = &Kg[(size_t)bb * S_ * kvstride + kvh * DK_];
  const bf16* vbase = &Vg[(size_t)bb * S_ * kvstride + kvh * DK_];

  // ---- stage Q tile [64,128] via async global->LDS (b128 = 8 bf16/lane) ----
  {
    const unsigned long long gq =
        (unsigned long long)(size_t)&Q[((size_t)(bb * S_ + qb * 64)) * (NQ_ * DK_) + h * DK_];
    const unsigned ldsbase = lds_off(&Qs[0][0]);
#pragma unroll
    for (int i = 0; i < 8; ++i) {
      int li = i * 128 + t;
      int r = li >> 4, c16 = li & 15;                    // row, 16-byte chunk
      unsigned laddr = ldsbase + (unsigned)((r * 128 + c16 * 8) * 2);
      unsigned goff  = (unsigned)((r * (NQ_ * DK_) + c16 * 8) * 2);
      asm volatile("global_load_async_to_lds_b128 %0, %1, %2"
                   :: "v"(laddr), "v"(goff), "s"(gq) : "memory");
    }
  }
  // kick off the first K/V tile DMA while the Q copies are in flight
  if (w == 0)
    tdm_issue_pair(kbase, vbase, lds_off(&Ks[0][0][0]), lds_off(&Vs[0][0][0]), g1);
  asm volatile("s_wait_asynccnt 0x0" ::: "memory");
  __syncthreads();

  // build Q A-fragments from LDS (4 chunks of K=32)
  AFrag qf[4];
#pragma unroll
  for (int c = 0; c < 4; ++c)
#pragma unroll
    for (int v = 0; v < 8; ++v) {
      int kk = c * 32 + hi * 16 + 2 * v;
      qf[c].p[v] = *(const bf16x2*)&Qs[w * 16 + lo][kk];
    }

  CFrag o[8] = {};
  float mrun[8], lrun[8];
#pragma unroll
  for (int i = 0; i < 8; ++i) { mrun[i] = -3.0e38f; lrun[i] = 0.f; }

  for (int kb = 0; kb <= qb; ++kb) {
    const int buf = kb & 1;
    __syncthreads();   // all waves done reading buf^1 from iteration kb-1
    if (w == 0) {
      if (kb < qb) {
        // prefetch next tile into the other buffer, then drain only the
        // older pair: TENSORcnt completes in order, <=2 leaves the new
        // pair's DMA running under this iteration's compute.
        tdm_issue_pair(kbase + (size_t)(kb + 1) * 64 * kvstride,
                       vbase + (size_t)(kb + 1) * 64 * kvstride,
                       lds_off(&Ks[buf ^ 1][0][0]), lds_off(&Vs[buf ^ 1][0][0]), g1);
        __builtin_amdgcn_s_wait_tensorcnt(2);
      } else {
        __builtin_amdgcn_s_wait_tensorcnt(0);
      }
    }
    __syncthreads();   // publish Ks[buf]/Vs[buf] to all waves

    // transpose V into VT[dk][key] for contiguous PV B-fragment reads
    for (int i = 0; i < 64; ++i) {
      int li = i * 128 + t;
      int r = li >> 7, c = li & 127;
      VT[c][r] = Vs[buf][r][c];
    }
    __syncthreads();

    // S = Q * K^T : 4 n-tiles of 16 keys, accumulate over DK in 4 chunks
    CFrag st[4];
#pragma unroll
    for (int nt = 0; nt < 4; ++nt) {
      v8f a = {};
#pragma unroll
      for (int c = 0; c < 4; ++c) {
        AFrag bf;
#pragma unroll
        for (int v = 0; v < 8; ++v) {
          int kk = c * 32 + hi * 16 + 2 * v;   // dk index (reduction dim)
          bf.p[v] = *(const bf16x2*)&Ks[buf][nt * 16 + lo][kk];
        }
        a = __builtin_amdgcn_wmma_f32_16x16x32_bf16(
            false, qf[c].v, false, bf.v, (short)0, a, false, false);
      }
      st[nt].v = a;
    }

    // scale + causal mask (only the diagonal block needs masking)
#pragma unroll
    for (int nt = 0; nt < 4; ++nt)
#pragma unroll
      for (int i = 0; i < 8; ++i) {
        float s = st[nt].e[i] * scale;
        if (kb == qb) {
          int qrow = w * 16 + hi * 8 + i;   // row within 64-block
          int kcol = nt * 16 + lo;          // key within 64-block
          if (kcol > qrow) s = -1.0e9f;
        }
        st[nt].e[i] = s;
      }

    // row max across 64 keys (4 regs + 16-lane half-wave reduction)
    float rmax[8];
#pragma unroll
    for (int i = 0; i < 8; ++i) {
      float v = fmaxf(fmaxf(st[0].e[i], st[1].e[i]), fmaxf(st[2].e[i], st[3].e[i]));
#pragma unroll
      for (int m = 1; m <= 8; m <<= 1) v = fmaxf(v, __shfl_xor(v, m, 32));
      rmax[i] = v;
    }
    // online rescale
#pragma unroll
    for (int i = 0; i < 8; ++i) {
      float mn = fmaxf(mrun[i], rmax[i]);
      float al = __expf(mrun[i] - mn);
      mrun[i] = mn;
      lrun[i] *= al;
#pragma unroll
      for (int dt = 0; dt < 8; ++dt) o[dt].e[i] *= al;
    }
    // P = exp(S - m), row sums
#pragma unroll
    for (int nt = 0; nt < 4; ++nt)
#pragma unroll
      for (int i = 0; i < 8; ++i) st[nt].e[i] = __expf(st[nt].e[i] - mrun[i]);
#pragma unroll
    for (int i = 0; i < 8; ++i) {
      float v = st[0].e[i] + st[1].e[i] + st[2].e[i] + st[3].e[i];
#pragma unroll
      for (int m = 1; m <= 8; m <<= 1) v += __shfl_xor(v, m, 32);
      lrun[i] += v;
    }
    // restage P (C layout -> row-major) for A-fragment reads
#pragma unroll
    for (int nt = 0; nt < 4; ++nt)
#pragma unroll
      for (int i = 0; i < 8; ++i)
        Ps[w][hi * 8 + i][nt * 16 + lo] = (bf16)st[nt].e[i];
    __syncthreads();

    // O += P * V : reduction over 64 keys in 2 chunks of 32
#pragma unroll
    for (int kc = 0; kc < 2; ++kc) {
      AFrag pa;
#pragma unroll
      for (int v = 0; v < 8; ++v) {
        int kk = kc * 32 + hi * 16 + 2 * v;
        pa.p[v] = *(const bf16x2*)&Ps[w][lo][kk];
      }
#pragma unroll
      for (int dt = 0; dt < 8; ++dt) {
        AFrag vb;
#pragma unroll
        for (int v = 0; v < 8; ++v) {
          int kk = kc * 32 + hi * 16 + 2 * v;   // key index (reduction dim)
          vb.p[v] = *(const bf16x2*)&VT[dt * 16 + lo][kk];
        }
        o[dt].v = __builtin_amdgcn_wmma_f32_16x16x32_bf16(
            false, pa.v, false, vb.v, (short)0, o[dt].v, false, false);
      }
    }
  }

  // normalize and store attention output (bf16, [B*S, NQ*DK])
#pragma unroll
  for (int i = 0; i < 8; ++i) {
    float inv = 1.0f / lrun[i];
#pragma unroll
    for (int dt = 0; dt < 8; ++dt) o[dt].e[i] *= inv;
  }
#pragma unroll
  for (int dt = 0; dt < 8; ++dt)
#pragma unroll
    for (int i = 0; i < 8; ++i) {
      int q = qb * 64 + w * 16 + hi * 8 + i;
      O[((size_t)(bb * S_ + q)) * (NQ_ * DK_) + h * DK_ + dt * 16 + lo] = (bf16)o[dt].e[i];
    }
}

// =====================================================================
// Launch
// =====================================================================
extern "C" void kernel_launch(void* const* d_in, const int* in_sizes, int n_in,
                              void* d_out, int out_size, void* d_ws, size_t ws_size,
                              hipStream_t stream) {
  const float* x  = (const float*)d_in[0];   // [B,S,D]
  const float* Wq = (const float*)d_in[1];   // [D, NQ*DK]
  const float* Wk = (const float*)d_in[2];   // [D, NKV*DK]
  const float* Wv = (const float*)d_in[3];   // [D, NKV*DK]
  const float* Wo = (const float*)d_in[4];   // [NQ*DK, D]
  float* out = (float*)d_out;                // [B,S,D]

  // workspace layout (bf16 intermediates)
  char* ws = (char*)d_ws;
  bf16* Qw = (bf16*)(ws);                                        // ROWS_ x 4096
  bf16* Kw = (bf16*)(ws + (size_t)ROWS_ * (NQ_ * DK_) * 2);      // ROWS_ x 512
  bf16* Vw = (bf16*)(ws + (size_t)ROWS_ * (NQ_ * DK_) * 2
                        + (size_t)ROWS_ * (NKV_ * DK_) * 2);     // ROWS_ x 512
  bf16* Aw = (bf16*)(ws + (size_t)ROWS_ * (NQ_ * DK_) * 2
                        + (size_t)ROWS_ * (NKV_ * DK_) * 4);     // ROWS_ x 4096

  dim3 blk(256);
  // projections: x[4096,4096] @ W
  gemm_wmma_bf16<float, float, bf16>
      <<<dim3((NQ_ * DK_) / 128, ROWS_ / 128), blk, 0, stream>>>(x, Wq, Qw, ROWS_, NQ_ * DK_, D_);
  gemm_wmma_bf16<float, float, bf16>
      <<<dim3((NKV_ * DK_) / 128, ROWS_ / 128), blk, 0, stream>>>(x, Wk, Kw, ROWS_, NKV_ * DK_, D_);
  gemm_wmma_bf16<float, float, bf16>
      <<<dim3((NKV_ * DK_) / 128, ROWS_ / 128), blk, 0, stream>>>(x, Wv, Vw, ROWS_, NKV_ * DK_, D_);

  // causal GQA attention
  gqa_flash_attn<<<dim3(S_ / 64, B_ * NQ_), dim3(128), 0, stream>>>(Qw, Kw, Vw, Aw);

  // output projection: attn[4096,4096] @ Wo -> fp32 out
  gemm_wmma_bf16<bf16, float, float>
      <<<dim3(D_ / 128, ROWS_ / 128), blk, 0, stream>>>(Aw, Wo, out, ROWS_, D_, NQ_ * DK_);
}
